// DeepFloorModel_61289183314244
// MI455X (gfx1250) — matrix operand order repaired and checked by reading, not compiled, not run on hardware
//
#include <hip/hip_runtime.h>
#include <math.h>

// ---------------- problem constants (from reference setup_inputs) ------------
#define VV      2
#define BB      8
#define SS      2048
#define DD      512
#define VOC     8192
#define NSTEPS  16
#define DECAYF  0.999f
#define EPSF    1e-6f

typedef __attribute__((ext_vector_type(16))) _Float16 v16h;
typedef __attribute__((ext_vector_type(8)))  _Float16 v8h;
typedef __attribute__((ext_vector_type(8)))  float    v8f;
typedef __attribute__((ext_vector_type(4)))  unsigned v4u;
typedef __attribute__((ext_vector_type(8)))  int      v8i;
typedef __attribute__((ext_vector_type(4)))  int      v4i;

// EPI: 0=none, 1=sigmoid, 2=exact gelu, 3=phi (elu+1)
template<int EPI>
__device__ __forceinline__ float apply_epi(float v)
{
    if (EPI == 1) return 1.f / (1.f + expf(-v));
    if (EPI == 2) return 0.5f * v * (1.f + erff(v * 0.70710678f));
    if (EPI == 3) return (v > 0.f) ? (v + 1.f) : expf(v);
    return v;
}

// ---------------------------------------------------------------------------
// TDM: DMA one 2D f16 tile (tileRows x tileK elements, row stride = strideElems)
// from global memory into LDS at byte offset ldsOff.  Descriptor per CDNA5 ISA
// "Tensor DMA Descriptor (D#)" (CDNA5 ISA Sec 8):
//   group0: [1:0]count=1 | [63:32]lds_addr | [120:64]global_addr | [127:126]type=2
//   group1: [17:16]data_size=1 (2B) | [79:48]tensor_dim0 | [111:80]tensor_dim1
//           | [127:112]tile_dim0 | [143:128]tile_dim1 | [207:160]dim0_stride
//   groups2/3: zero (2D tensor)
// Tracked by TENSORcnt; completion waited with s_wait_tensorcnt.
// ---------------------------------------------------------------------------
__device__ __forceinline__ void tdm_load_tile(const _Float16* gptr, unsigned ldsOff,
                                              int tileK, int tileRows, long long strideElems)
{
    unsigned long long ga = (unsigned long long)(size_t)gptr;
    unsigned long long st = (unsigned long long)strideElems;
    unsigned td0 = (unsigned)strideElems;      // tensor_dim0 >= tile width (no OOB)
    unsigned td1 = (unsigned)tileRows;

    v4u g0;
    g0.x = 1u;                                             // count=1, user mode
    g0.y = ldsOff;                                         // lds_addr (bytes)
    g0.z = (unsigned)(ga & 0xFFFFFFFFu);                   // global_addr[31:0]
    g0.w = (unsigned)((ga >> 32) & 0x01FFFFFFu) | (2u << 30); // addr[56:32] | type=2

    v8i g1;
    g1[0] = (int)(1u << 16);                               // data_size=1 (2 bytes)
    g1[1] = (int)((td0 & 0xFFFFu) << 16);                  // tensor_dim0[15:0] @bit48
    g1[2] = (int)(((td0 >> 16) & 0xFFFFu) | ((td1 & 0xFFFFu) << 16));
    g1[3] = (int)(((td1 >> 16) & 0xFFFFu) | (((unsigned)tileK & 0xFFFFu) << 16));
    g1[4] = (int)((unsigned)tileRows & 0xFFFFu);           // tile_dim1 ; tile_dim2=0
    g1[5] = (int)(st & 0xFFFFFFFFu);                       // dim0_stride[31:0]
    g1[6] = (int)((st >> 32) & 0xFFFFu);                   // dim0_stride[47:32]
    g1[7] = 0;

    v4i z = {0, 0, 0, 0};
#if __clang_major__ >= 23
    v8i z8 = {0, 0, 0, 0, 0, 0, 0, 0};
    __builtin_amdgcn_tensor_load_to_lds(g0, g1, z, z, z8, 0);
#else
    __builtin_amdgcn_tensor_load_to_lds(g0, g1, z, z, 0);
#endif
}

// ---------------------------------------------------------------------------
// Batched WMMA GEMM:  C = epi( scale * (A @ B) )
//   A : f16 [M,K] row-major          elem(m,k) = A [m*lda + k]
//   BT: f16 B transposed, [N,K]      elem(k,n) = BT[n*ldb + k]
//   C : f32/f16; STORE_T stores C transposed: CT[n*ldc + m] (vectorized v8h)
// Block: 128 thr = 4 waves, block tile 64x64, wave tile 32x32 = 2x2 WMMA
// tiles via v_wmma_f32_16x16x32_f16.  Tiles staged by the Tensor Data Mover
// (tensor_load_to_lds, double-buffered, s_wait_tensorcnt) -- wave 0 drives the
// DMA while all 4 waves compute.  All M,N used are multiples of 64 and K of
// 32 -> no bounds checks -> EXEC all-ones around WMMA (ISA requirement).
// ---------------------------------------------------------------------------
template<int EPI, bool STORE_HALF, bool STORE_T>
__global__ __launch_bounds__(128)
void gemm_wmma(const _Float16* __restrict__ Ag, const _Float16* __restrict__ BTg,
               void* __restrict__ Cg,
               int K, int lda, int ldb, int ldc,
               long long sA, long long sB, long long sC, float scale)
{
    __shared__ _Float16 As[2][64][32];   // [buf][m][k]
    __shared__ _Float16 Bs[2][64][32];   // [buf][n][k]  (B^T tile)

    const long long mBase = (long long)blockIdx.y * 64;
    const long long nBase = (long long)blockIdx.x * 64;

    const _Float16* A  = Ag  + (long long)blockIdx.z * sA + mBase * lda;
    const _Float16* BT = BTg + (long long)blockIdx.z * sB + nBase * ldb;

    const int lane = threadIdx.x;
    const int wave = threadIdx.y;          // 0..3
    const int wm   = (wave >> 1) * 32;     // wave M offset in block tile
    const int wn   = (wave & 1) * 32;      // wave N offset
    const int grp  = lane >> 4;            // 0: lanes 0-15, 1: lanes 16-31
    const int l16  = lane & 15;

    const bool loader = (wave == 0);       // wave-uniform branch
    const int  nCh    = K / 32;

    if (loader) {   // preload chunk 0 into buffer 0
        tdm_load_tile(A,  (unsigned)(size_t)(void*)&As[0][0][0], 32, 64, lda);
        tdm_load_tile(BT, (unsigned)(size_t)(void*)&Bs[0][0][0], 32, 64, ldb);
    }

    v8f acc[2][2] = {};

    for (int i = 0; i < nCh; ++i) {
        const int buf = i & 1;
        if (loader) {
            __builtin_amdgcn_s_wait_tensorcnt(0);   // chunk i landed in LDS
            if (i + 1 < nCh) {                      // DMA chunk i+1 into other buf
                tdm_load_tile(A  + (i + 1) * 32,
                              (unsigned)(size_t)(void*)&As[buf ^ 1][0][0], 32, 64, lda);
                tdm_load_tile(BT + (i + 1) * 32,
                              (unsigned)(size_t)(void*)&Bs[buf ^ 1][0][0], 32, 64, ldb);
            }
        }
        __syncthreads();   // tile[buf] visible to all waves

        // Fragments per ISA lane layouts:
        //  A 16x32: lanes 0-15 K={0..7,16..23}, lanes 16-31 K={8..15,24..31}
        //  B 32x16: lanes 0-15 K={0..7,16..23}, lanes 16-31 K={8..15,24..31}
        v16h af[2], bf[2];
        #pragma unroll
        for (int t = 0; t < 2; ++t) {
            int ar = wm + t * 16 + l16;
            v8h alo = *(const v8h*)&As[buf][ar][grp * 8];
            v8h ahi = *(const v8h*)&As[buf][ar][16 + grp * 8];
            int br = wn + t * 16 + l16;
            v8h blo = *(const v8h*)&Bs[buf][br][grp * 8];
            v8h bhi = *(const v8h*)&Bs[buf][br][16 + grp * 8];
            #pragma unroll
            for (int j = 0; j < 8; ++j) {
                af[t][j] = alo[j]; af[t][j + 8] = ahi[j];
                bf[t][j] = blo[j]; bf[t][j + 8] = bhi[j];
            }
        }
        #pragma unroll
        for (int mi = 0; mi < 2; ++mi)
            #pragma unroll
            for (int ni = 0; ni < 2; ++ni)
                acc[mi][ni] = __builtin_amdgcn_wmma_f32_16x16x32_f16(
                    false, af[mi], false, bf[ni], (short)0,
                    acc[mi][ni], false, false);
        __syncthreads();   // done reading tile[buf] before DMA overwrites it
    }

    // epilogue: f32 C layout -> VGPR r holds M = r + 8*grp, N = lane&15
    #pragma unroll
    for (int mi = 0; mi < 2; ++mi) {
        #pragma unroll
        for (int ni = 0; ni < 2; ++ni) {
            long long gmb = mBase + wm + mi * 16 + 8 * grp;   // + r (0..7)
            long long gn  = nBase + wn + ni * 16 + l16;
            if (STORE_T && STORE_HALF) {
                v8h o;
                #pragma unroll
                for (int r = 0; r < 8; ++r)
                    o[r] = (_Float16)apply_epi<EPI>(acc[mi][ni][r] * scale);
                *(v8h*)&((_Float16*)Cg)[(long long)blockIdx.z * sC + gn * ldc + gmb] = o;
            } else {
                #pragma unroll
                for (int r = 0; r < 8; ++r) {
                    float v = apply_epi<EPI>(acc[mi][ni][r] * scale);
                    long long ci = (long long)blockIdx.z * sC + (gmb + r) * ldc + gn;
                    if (STORE_HALF) ((_Float16*)Cg)[ci] = (_Float16)v;
                    else            ((float*)Cg)[ci]    = v;
                }
            }
        }
    }
}

// ---------------- RMSNorm one row (D=512) -> f16 ---------------------------
__global__ __launch_bounds__(256)
void rmsnorm_to_h(const float* __restrict__ x, _Float16* __restrict__ y)
{
    __shared__ float red[256];
    long long base = (long long)blockIdx.x * DD;
    int t = threadIdx.x;
    float a = x[base + t];
    float b = x[base + t + 256];
    red[t] = a * a + b * b;
    __syncthreads();
    #pragma unroll
    for (int s = 128; s > 0; s >>= 1) {
        if (t < s) red[t] += red[t + s];
        __syncthreads();
    }
    float sc = rsqrtf(red[0] * (1.f / (float)DD) + EPSF);
    y[base + t]       = (_Float16)(a * sc);
    y[base + t + 256] = (_Float16)(b * sc);
}

// ---------------- elementwise helpers --------------------------------------
// W [Kd][Nd] f32  ->  WT [Nd][Kd] f16   (grid covers Kd*Nd exactly)
__global__ void transpose_to_h_k(const float* __restrict__ Wm,
                                 _Float16* __restrict__ WT, int Kd, int Nd)
{
    long long g = (long long)blockIdx.x * blockDim.x + threadIdx.x;
    int       n = (int)(g % Nd);
    long long k = g / Nd;
    WT[(long long)n * Kd + k] = (_Float16)Wm[g];
}

__global__ void gather_embed_k(const int* __restrict__ inp,
                               const float* __restrict__ emb,
                               float* __restrict__ st)
{
    long long g = (long long)blockIdx.x * blockDim.x + threadIdx.x; // exact cover
    int d        = (int)(g % DD);
    long long r  = g / DD;
    int s        = (int)(r % SS);
    int vb       = (int)(r / SS);
    int v        = vb / BB;
    int b        = vb - v * BB;
    int tok      = inp[(long long)b * SS + s];
    st[g] = emb[((long long)v * VOC + tok) * DD + d];
}

__global__ void gated_add_k(float* __restrict__ st, const float* __restrict__ g,
                            const float* __restrict__ c)
{
    long long i = (long long)blockIdx.x * blockDim.x + threadIdx.x;
    st[i] += g[i] * c[i];
}

__global__ void add_inplace_k(float* __restrict__ st, const float* __restrict__ x)
{
    long long i = (long long)blockIdx.x * blockDim.x + threadIdx.x;
    st[i] += x[i];
}

// acc = DECAY*acc + step_acc (inv_s folded into GEMM scale);
// also writes transposed f16 copy: accT[e][d] = acc[d][e]  (B^T for q@acc)
__global__ void acc_update_k(float* __restrict__ acc,
                             const float* __restrict__ sacc,
                             _Float16* __restrict__ accT)
{
    long long i = (long long)blockIdx.x * blockDim.x + threadIdx.x;
    long long local = i % ((long long)DD * DD);
    long long vb    = i / ((long long)DD * DD);
    int d = (int)(local / DD);
    int e = (int)(local % DD);
    float a = DECAYF * acc[i] + sacc[i];
    acc[i] = a;
    accT[vb * DD * DD + (long long)e * DD + d] = (_Float16)a;
}

__global__ void zero_f32_k(float* __restrict__ p)
{
    long long i = (long long)blockIdx.x * blockDim.x + threadIdx.x;
    p[i] = 0.f;
}

__global__ void mean_combine_k(const float* __restrict__ st,
                               float* __restrict__ comb, long long off)
{
    long long i = (long long)blockIdx.x * blockDim.x + threadIdx.x;
    comb[i] = 0.5f * (st[i] + st[i + off]);
}

// ---------------------------------------------------------------------------
extern "C" void kernel_launch(void* const* d_in, const int* in_sizes, int n_in,
                              void* d_out, int out_size, void* d_ws, size_t ws_size,
                              hipStream_t stream)
{
    const int*   inputs = (const int*)  d_in[0];
    const float* emb    = (const float*)d_in[1];
    const float* Wg     = (const float*)d_in[2];
    const float* Wu     = (const float*)d_in[3];
    const float* Wd     = (const float*)d_in[4];
    const float* Wq     = (const float*)d_in[5];
    const float* Wk     = (const float*)d_in[6];
    const float* Wv     = (const float*)d_in[7];
    const float* Wo     = (const float*)d_in[8];
    const float* Wlm    = (const float*)d_in[9];
    float* out = (float*)d_out;

    const long long VB   = (long long)VV * BB;          // 16
    const int       MT   = (int)(VB * SS);              // 32768 fused rows
    const long long NSD  = VB * SS * DD;                // 16,777,216
    const long long NDD  = VB * DD * DD;                // 4,194,304
    const long long NBSD = (long long)BB * SS * DD;     // 8,388,608

    // ------- workspace carve-out (bytes) -------
    char* w = (char*)d_ws;
    size_t off = 0;
    auto carve = [&](size_t bytes) { void* p = w + off; off += (bytes + 255) & ~(size_t)255; return p; };
    float*    state  = (float*)   carve(NSD * 4);
    _Float16* n_h    = (_Float16*)carve(NSD * 2);
    float*    gate   = (float*)   carve(NSD * 4);
    _Float16* u_h    = (_Float16*)carve(NSD * 2 * 2);   // [MT, 2D]
    float*    cand   = (float*)   carve(NSD * 4);       // also reused for Wo delta
    _Float16* q_h    = (_Float16*)carve(NSD * 2);       // [MT, D]
    _Float16* k_T    = (_Float16*)carve(NSD * 2);       // [D, MT] (transposed)
    _Float16* v_T    = (_Float16*)carve(NSD * 2);       // [D, MT] (transposed)
    _Float16* mix_h  = (_Float16*)carve(NSD * 2);       // [MT, D]
    float*    sacc   = (float*)   carve(NDD * 4);
    float*    acc    = (float*)   carve(NDD * 4);
    _Float16* accT   = (_Float16*)carve(NDD * 2);       // [vb][e][d]
    float*    comb   = (float*)   carve(NBSD * 4);
    _Float16* comb_h = (_Float16*)carve(NBSD * 2);
    _Float16* WgT    = (_Float16*)carve((size_t)DD * DD * 2);       // [D, D]
    _Float16* WuT    = (_Float16*)carve((size_t)DD * 2 * DD * 2);   // [2D, D]
    _Float16* WdT    = (_Float16*)carve((size_t)2 * DD * DD * 2);   // [D, 2D]
    _Float16* WqT    = (_Float16*)carve((size_t)DD * DD * 2);
    _Float16* WkT    = (_Float16*)carve((size_t)DD * DD * 2);
    _Float16* WvT    = (_Float16*)carve((size_t)DD * DD * 2);
    _Float16* WoT    = (_Float16*)carve((size_t)DD * DD * 2);
    _Float16* WlmT   = (_Float16*)carve((size_t)DD * VOC * 2);      // [VOC, D]

    const dim3 gblk(32, 4);                  // 4 waves / block, 64x64 tile
    const float inv_s  = 1.f / (float)SS;
    const float inv_sd = 1.f / sqrtf((float)DD);

    // ------- one-time weight transpose+convert, embed gather, acc=0 -------
    auto trn = [&](const float* s, _Float16* d, int Kd, int Nd) {
        long long n = (long long)Kd * Nd;
        transpose_to_h_k<<<(unsigned)(n / 256), 256, 0, stream>>>(s, d, Kd, Nd);
    };
    trn(Wg,  WgT,  DD, DD);
    trn(Wu,  WuT,  DD, 2 * DD);
    trn(Wd,  WdT,  2 * DD, DD);
    trn(Wq,  WqT,  DD, DD);
    trn(Wk,  WkT,  DD, DD);
    trn(Wv,  WvT,  DD, DD);
    trn(Wo,  WoT,  DD, DD);
    trn(Wlm, WlmT, DD, VOC);

    gather_embed_k<<<(unsigned)(NSD / 256), 256, 0, stream>>>(inputs, emb, state);
    zero_f32_k<<<(unsigned)(NDD / 256), 256, 0, stream>>>(acc);

    // ------- 16 recurrence steps -------
    for (int step = 0; step < NSTEPS; ++step) {
        // n = rms(state)
        rmsnorm_to_h<<<(unsigned)(VB * SS), 256, 0, stream>>>(state, n_h);
        // gate = sigmoid(n @ Wg)                           [MT, D] f32
        gemm_wmma<1, false, false><<<dim3(DD / 64, MT / 64, 1), gblk, 0, stream>>>(
            n_h, WgT, gate, DD, DD, DD, DD, 0, 0, 0, 1.f);
        // u = gelu(n @ Wu)                                 [MT, 2D] f16
        gemm_wmma<2, true, false><<<dim3(2 * DD / 64, MT / 64, 1), gblk, 0, stream>>>(
            n_h, WuT, u_h, DD, DD, DD, 2 * DD, 0, 0, 0, 1.f);
        // cand = u @ Wd                                    [MT, D] f32
        gemm_wmma<0, false, false><<<dim3(DD / 64, MT / 64, 1), gblk, 0, stream>>>(
            u_h, WdT, cand, 2 * DD, 2 * DD, 2 * DD, DD, 0, 0, 0, 1.f);
        // state += gate * cand
        gated_add_k<<<(unsigned)(NSD / 256), 256, 0, stream>>>(state, gate, cand);

        // n2 = rms(state)
        rmsnorm_to_h<<<(unsigned)(VB * SS), 256, 0, stream>>>(state, n_h);
        // q = phi(n2 @ Wq) [MT,D] f16 ; k,v stored TRANSPOSED [D,MT] f16
        gemm_wmma<3, true, false><<<dim3(DD / 64, MT / 64, 1), gblk, 0, stream>>>(
            n_h, WqT, q_h, DD, DD, DD, DD, 0, 0, 0, 1.f);
        gemm_wmma<3, true, true><<<dim3(DD / 64, MT / 64, 1), gblk, 0, stream>>>(
            n_h, WkT, k_T, DD, DD, DD, MT, 0, 0, 0, 1.f);
        gemm_wmma<0, true, true><<<dim3(DD / 64, MT / 64, 1), gblk, 0, stream>>>(
            n_h, WvT, v_T, DD, DD, DD, MT, 0, 0, 0, 1.f);

        // step_acc[vb] = (k^T @ v)/S : A = k_T rows=D cols=S(vb), BT = v_T
        gemm_wmma<0, false, false><<<dim3(DD / 64, DD / 64, (unsigned)VB), gblk, 0, stream>>>(
            k_T, v_T, sacc, SS, MT, MT, DD,
            (long long)SS, (long long)SS, (long long)DD * DD, inv_s);
        // acc = DECAY*acc + step_acc ; accT = f16(acc)^T
        acc_update_k<<<(unsigned)(NDD / 256), 256, 0, stream>>>(acc, sacc, accT);

        // mixed = (q @ acc)/sqrt(D)  : batched, BT = accT, f16 out [MT, D]
        gemm_wmma<0, true, false><<<dim3(DD / 64, SS / 64, (unsigned)VB), gblk, 0, stream>>>(
            q_h, accT, mix_h, DD, DD, DD, DD,
            (long long)SS * DD, (long long)DD * DD, (long long)SS * DD, inv_sd);
        // delta = mixed @ Wo ; state += delta
        gemm_wmma<0, false, false><<<dim3(DD / 64, MT / 64, 1), gblk, 0, stream>>>(
            mix_h, WoT, cand, DD, DD, DD, DD, 0, 0, 0, 1.f);
        add_inplace_k<<<(unsigned)(NSD / 256), 256, 0, stream>>>(state, cand);
    }

    // ------- combine views, final norm + LM head -------
    mean_combine_k<<<(unsigned)(NBSD / 256), 256, 0, stream>>>(state, comb, NBSD);
    rmsnorm_to_h<<<(unsigned)(BB * SS), 256, 0, stream>>>(comb, comb_h);
    gemm_wmma<0, false, false><<<dim3(VOC / 64, (BB * SS) / 64, 1), gblk, 0, stream>>>(
        comb_h, WlmT, out, DD, DD, DD, VOC, 0, 0, 0, 1.f);

    (void)in_sizes; (void)n_in; (void)out_size; (void)ws_size;
}
